// LSTMModel_59038620450849
// MI455X (gfx1250) — compile-verified, matrix operand order checked by reading, not run on hardware
//
#include <hip/hip_runtime.h>

typedef __attribute__((ext_vector_type(16))) _Float16 v16h;
typedef __attribute__((ext_vector_type(8)))  _Float16 v8h;
typedef __attribute__((ext_vector_type(8)))  float    v8f;
typedef __attribute__((ext_vector_type(2)))  float    v2f;

#define T_LEN   2048
#define H_DIM   32
#define LDS_PITCH 40   // halves per row (80B) -> 16B aligned, spreads LDS banks

#if __has_builtin(__builtin_amdgcn_wmma_f32_16x16x4_f32)
#define HAVE_X4 1
#else
#define HAVE_X4 0
#endif

#if __has_builtin(__builtin_amdgcn_tanhf)
#define HAVE_TANH 1
#else
#define HAVE_TANH 0
#endif

__device__ __forceinline__ float fast_tanh(float x) {
#if HAVE_TANH
    return __builtin_amdgcn_tanhf(x);            // v_tanh_f32: 1 trans op
#else
    return 1.0f - 2.0f * __builtin_amdgcn_rcpf(__expf(2.0f * x) + 1.0f);
#endif
}
__device__ __forceinline__ float fast_sigmoid(float x) {
#if HAVE_TANH
    return fmaf(0.5f, __builtin_amdgcn_tanhf(0.5f * x), 0.5f);
#else
    return __builtin_amdgcn_rcpf(1.0f + __expf(-x));
#endif
}

// 2 waves per WG share one 16-row batch tile.
// Wave w owns gate n-columns [16w, 16w+16) -> gate tiles {w, 2+w, 4+w, 6+w}
// (gates: i=[0,32) f=[32,64) g=[64,96) o=[96,128)) and its half of c/h state.
// Both waves rebuild the full 16x32 f16 A operand (h) from a shared LDS tile.
__global__ void __launch_bounds__(64)
lstm_scan_wmma(const float* __restrict__ x,      // [B, T, 3]
               const float* __restrict__ Wih,    // [128, 3]
               const float* __restrict__ Whh,    // [128, 32]
               const float* __restrict__ bih,    // [128]
               const float* __restrict__ bhh,    // [128]
               float* __restrict__ out)          // [B, 32]
{
    const int lane  = threadIdx.x & 31;  // 0..31
    const int wid   = threadIdx.x >> 5;  // 0..1 : n-half owned by this wave
    const int ln    = lane & 15;         // n (B/C/D cols) or m (A rows)
    const int hl    = lane >> 4;         // half select within the wave
    const int bbase = blockIdx.x * 16;   // 16 batch rows per WG

    // Gate tiles owned by this wave: i, f, g, o at n-half `wid`
    // tile index = 2*q + wid for q = 0(i),1(f),2(g),3(o)

    // ---- W_hh as f16 B-matrices (32x16 per owned tile), resident in VGPRs ----
    // B layout: lane n=ln ; element j<8 -> k=8*hl+j ; j>=8 -> k=16+8*hl+(j-8)
    v16h whh[4];
#pragma unroll
    for (int q = 0; q < 4; ++q) {
        const int tile = 2 * q + wid;
        const float* wr = Whh + (tile * 16 + ln) * H_DIM;  // row g of W_hh
        v16h w;
#pragma unroll
        for (int j = 0; j < 16; ++j) {
            const int k = (j < 8) ? (8 * hl + j) : (16 + 8 * hl + (j - 8));
            w[j] = (_Float16)wr[k];
        }
        whh[q] = w;
    }

    // ---- W_ih (+bias) for the x-contribution ----
#if HAVE_X4
    // f32 B-matrix 4x16 per tile: lane n=ln; hl==0 -> rows K0,K1 ; hl==1 -> rows K2,K3(=bias)
    v2f wib[4];
#pragma unroll
    for (int q = 0; q < 4; ++q) {
        const int ng = (2 * q + wid) * 16 + ln;
        v2f w;
        if (hl == 0) { w.x = Wih[ng * 3 + 0]; w.y = Wih[ng * 3 + 1]; }
        else         { w.x = Wih[ng * 3 + 2]; w.y = bih[ng] + bhh[ng]; }
        wib[q] = w;
    }
#else
    float4 wib4[4];
#pragma unroll
    for (int q = 0; q < 4; ++q) {
        const int ng = (2 * q + wid) * 16 + ln;
        wib4[q] = make_float4(Wih[ng * 3 + 0], Wih[ng * 3 + 1], Wih[ng * 3 + 2],
                              bih[ng] + bhh[ng]);
    }
#endif

    // ---- state (this wave's n-half only) ----
    v8f  cst = {};             // cell state, D-layout, n = ln + 16*wid
    v8f  hst = {};             // hidden state (f32 copy for final output)
    v16h hA  = {};             // full h as f16 WMMA A operand (16x32), starts at 0

    __shared__ __align__(16) _Float16 hbuf[16 * LDS_PITCH];

    // each lane's x row (A-layout row m = ln); both waves read the same rows (L0-cached)
    const float* xrow = x + (size_t)(bbase + ln) * T_LEN * 3;

    for (int t = 0; t < T_LEN; ++t) {
        if (t + 16 < T_LEN) __builtin_prefetch(xrow + (size_t)(t + 16) * 3, 0, 1);

#if HAVE_X4
        // A = [x0 x1 x2 1] 16x4 f32 : lane m=ln; hl==0 -> (K0,K1)=(x0,x1) ; hl==1 -> (K2,K3)=(x2,1)
        v2f ax;
        ax.x = xrow[t * 3 + 2 * hl];                       // hl0: x0 ; hl1: x2
        float a1 = xrow[t * 3 + 1];                        // in-bounds for all lanes
        ax.y = hl ? 1.0f : a1;
#else
        const float xr0 = xrow[t * 3 + 0];
        const float xr1 = xrow[t * 3 + 1];
        const float xr2 = xrow[t * 3 + 2];
        float bx0[8], bx1[8], bx2[8];
#pragma unroll
        for (int v = 0; v < 8; ++v) {      // row m = v + 8*hl is held by lane (v+8*hl)
            const int src = v + 8 * hl;
            bx0[v] = __shfl(xr0, src, 32);
            bx1[v] = __shfl(xr1, src, 32);
            bx2[v] = __shfl(xr2, src, 32);
        }
#endif

        // gates for one owned 16-wide n-tile: D = hA*Whh + (x*Wih + bias)
        auto gate = [&](int q) -> v8f {
#if HAVE_X4
            v8f z = {};
            v8f cin = __builtin_amdgcn_wmma_f32_16x16x4_f32(
                false, ax, false, wib[q], (short)0, z, false, false);
#else
            v8f cin;
#pragma unroll
            for (int e = 0; e < 8; ++e)
                cin[e] = wib4[q].w + bx0[e] * wib4[q].x +
                         bx1[e] * wib4[q].y + bx2[e] * wib4[q].z;
#endif
            return __builtin_amdgcn_wmma_f32_16x16x32_f16(
                false, hA, false, whh[q], (short)0, cin, false, false);
        };

        {
            v8f gi = gate(0);
            v8f gf = gate(1);
            v8f gg = gate(2);
            v8f go = gate(3);
            v8f hn;
#pragma unroll
            for (int e = 0; e < 8; ++e) {
                const float iv = fast_sigmoid(gi[e]);
                const float fv = fast_sigmoid(gf[e]);
                const float gv = fast_tanh(gg[e]);
                const float ov = fast_sigmoid(go[e]);
                const float cn = fv * cst[e] + iv * gv;
                cst[e] = cn;
                hn[e]  = ov * fast_tanh(cn);
            }
            hst = hn;
            // D-layout -> LDS f16 tile: row m = e + 8*hl, col n = ln + 16*wid
#pragma unroll
            for (int e = 0; e < 8; ++e)
                hbuf[(e + 8 * hl) * LDS_PITCH + (ln + 16 * wid)] = (_Float16)hn[e];
        }
        __syncthreads();   // wave1's cols visible to wave0 and vice versa

        // A-layout reload (full 16x32): lane m=ln, two contiguous 8-half runs
        const v8h lo = *(const v8h*)&hbuf[ln * LDS_PITCH + 8 * hl];
        const v8h hi = *(const v8h*)&hbuf[ln * LDS_PITCH + 16 + 8 * hl];
        hA = __builtin_shufflevector(lo, hi, 0, 1, 2, 3, 4, 5, 6, 7,
                                     8, 9, 10, 11, 12, 13, 14, 15);
        __syncthreads();   // WAR guard: next step's stores vs this step's loads
    }

    // final h (f32 path, never rounded to f16): out[b][n], this wave's n-half
#pragma unroll
    for (int e = 0; e < 8; ++e)
        out[(size_t)(bbase + e + 8 * hl) * H_DIM + (ln + 16 * wid)] = hst[e];
}

extern "C" void kernel_launch(void* const* d_in, const int* in_sizes, int n_in,
                              void* d_out, int out_size, void* d_ws, size_t ws_size,
                              hipStream_t stream) {
    (void)in_sizes; (void)n_in; (void)d_ws; (void)ws_size; (void)out_size;
    const float* x   = (const float*)d_in[0];
    const float* Wih = (const float*)d_in[1];
    const float* Whh = (const float*)d_in[2];
    const float* bih = (const float*)d_in[3];
    const float* bhh = (const float*)d_in[4];
    float* out = (float*)d_out;

    const int B = 4096;
    lstm_scan_wmma<<<B / 16, 64, 0, stream>>>(x, Wih, Whh, bih, bhh, out);
}
